// Attn_Head_22926535426495
// MI455X (gfx1250) — matrix operand order repaired and checked by reading, not compiled, not run on hardware
//
#include <hip/hip_runtime.h>
#include <hip/hip_bf16.h>
#include <stdint.h>

// GAT attention head, fused: proj -> online softmax stats -> bf16-WMMA aggregation.
// N=8192, F=512, H=64, fp32 in/out, bf16 tensor-core path for coefs @ fts.

#define NN 8192
#define FF 512
#define HH 64

typedef __attribute__((ext_vector_type(16))) __bf16        v16bf;
typedef __attribute__((ext_vector_type(2)))  __bf16        v2bf;
typedef __attribute__((ext_vector_type(8)))  float         v8f;
typedef __attribute__((ext_vector_type(4)))  float         v4f;
typedef __attribute__((ext_vector_type(8)))  unsigned int  v8u;
typedef __attribute__((ext_vector_type(4)))  unsigned int  v4u;

__device__ __forceinline__ unsigned int pack2_bf16(float lo, float hi) {
#if __has_builtin(__builtin_amdgcn_cvt_pk_bf16_f32)
    v2bf p = __builtin_amdgcn_cvt_pk_bf16_f32(lo, hi);
    return __builtin_bit_cast(unsigned int, p);
#else
    // round-to-nearest-even f32->bf16, packed pair {lo[15:0], hi[31:16]}
    unsigned int a = __builtin_bit_cast(unsigned int, lo);
    unsigned int b = __builtin_bit_cast(unsigned int, hi);
    a += 0x7FFFu + ((a >> 16) & 1u);
    b += 0x7FFFu + ((b >> 16) & 1u);
    return (a >> 16) | (b & 0xFFFF0000u);
#endif
}

// ---------------------------------------------------------------------------
// Kernel A: fts = features @ W (fp32 accum), store fts transposed as bf16
// [H][N] for WMMA-B loads, and f1[n] = fts.a1 + b1, f2[n] = fts.a2 + b2.
// Block: 256 threads = 4 nodes x 64 h. Grid: N/4 = 2048.
// ---------------------------------------------------------------------------
__global__ void proj_kernel(const float* __restrict__ features,
                            const float* __restrict__ W,
                            const float* __restrict__ a1, const float* __restrict__ b1,
                            const float* __restrict__ a2, const float* __restrict__ b2,
                            unsigned short* __restrict__ ftsT,
                            float* __restrict__ f1, float* __restrict__ f2) {
    __shared__ float sfeat[4 * FF];   // 8 KB
    __shared__ float red1[256];
    __shared__ float red2[256];
    const int tid = threadIdx.x;
    const long base = (long)blockIdx.x * 4 * FF;
#pragma unroll
    for (int j = 0; j < 8; ++j) {
        int idx = tid + j * 256;
        sfeat[idx] = features[base + idx];
    }
    __syncthreads();
    const int nl = tid >> 6;          // node within block
    const int h  = tid & 63;
    const int n  = blockIdx.x * 4 + nl;
    float acc = 0.f;
    const float* fr = &sfeat[nl * FF];
#pragma unroll 8
    for (int k = 0; k < FF; ++k)
        acc = fmaf(fr[k], W[k * HH + h], acc);   // W reads coalesced across h
    // transposed bf16 store (L2-resident, 1 MB total)
    ftsT[(long)h * NN + n] = (unsigned short)(pack2_bf16(acc, 0.f) & 0xFFFFu);
    // f1/f2 reductions over the 64 h-threads of this node
    red1[tid] = acc * a1[h];
    red2[tid] = acc * a2[h];
    __syncthreads();
    for (int off = 32; off > 0; off >>= 1) {
        if (h < off) { red1[tid] += red1[tid + off]; red2[tid] += red2[tid + off]; }
        __syncthreads();
    }
    if (h == 0) { f1[n] = red1[tid] + b1[0]; f2[n] = red2[tid] + b2[0]; }
}

// ---------------------------------------------------------------------------
// Kernel B: per-row online softmax stats over x = relu(f1[m]+f2[n]) + bias.
// Block: 256 threads = 4 rows x 64 cols (strided). Grid: N/4 = 2048.
// Streams bias with non-temporal loads (one full 256 MB pass).
// ---------------------------------------------------------------------------
__global__ void softstat_kernel(const float* __restrict__ bias,
                                const float* __restrict__ f1,
                                const float* __restrict__ f2,
                                float* __restrict__ mx_out,
                                float* __restrict__ rs_out) {
    __shared__ float smx[256], ssm[256];
    const int tid = threadIdx.x;
    const int r = tid >> 6, c0 = tid & 63;
    const int m = blockIdx.x * 4 + r;
    const float f1m = f1[m];
    const float* br = bias + (long)m * NN;
    float mx = -3.0e38f, sm = 0.f;
#pragma unroll 4
    for (int i = 0; i < NN / 64; ++i) {
        int c = c0 + i * 64;
        float x = fmaxf(f1m + f2[c], 0.f) + __builtin_nontemporal_load(br + c);
        float nm = fmaxf(mx, x);
        sm = sm * __expf(mx - nm) + __expf(x - nm);   // online rescale
        mx = nm;
    }
    smx[tid] = mx; ssm[tid] = sm;
    __syncthreads();
    for (int off = 32; off > 0; off >>= 1) {
        if (c0 < off) {
            float m2 = smx[tid + off], s2 = ssm[tid + off];
            float nm = fmaxf(smx[tid], m2);
            ssm[tid] = ssm[tid] * __expf(smx[tid] - nm) + s2 * __expf(m2 - nm);
            smx[tid] = nm;
        }
        __syncthreads();
    }
    if (c0 == 0) { mx_out[m] = smx[tid]; rs_out[m] = 1.0f / ssm[tid]; }
}

// ---------------------------------------------------------------------------
// Kernel C: out = elu( (softmax + bias) @ fts ) via v_wmma_f32_16x16x32_bf16.
// One block (8 waves) per 16-row tile; each wave owns a K-slice of 1024
// neighbors and 4 C-tiles (H=64 = 4 x 16). Coefs are computed in registers
// from a second streaming bias pass and packed straight into the bf16 A-matrix
// lane layout; B comes from bf16 ftsT (L2-hot). Partial C reduced in LDS.
// A layout (16-bit 16x32): lane-half K blocks {0-7}/{8-15} then {16-23}/{24-31}.
// B layout (16-bit 32x16): lanes 0-15 hold K=0..7,16..23, lanes 16-31 hold K=8..15,24..31.
// Grid: N/16 = 512.
// ---------------------------------------------------------------------------
__global__ void attn_agg_kernel(const float* __restrict__ bias,
                                const float* __restrict__ f1,
                                const float* __restrict__ f2,
                                const float* __restrict__ mx,
                                const float* __restrict__ rs,
                                const unsigned short* __restrict__ ftsT,
                                float* __restrict__ out) {
    __shared__ float cred[8 * 16 * HH];   // 32 KB: per-wave partial C tiles
    const int tid  = threadIdx.x;
    const int wave = tid >> 5;
    const int lane = tid & 31;
    const int m0   = blockIdx.x * 16;
    const int col  = lane & 15;           // M for A/C, N for B
    const int akoff = (lane >> 4) * 8;    // A: upper half-wave = K+8 (interleaved)
    const int bkoff = (lane >> 4) * 8;    // B: upper half-wave = K+8 (interleaved, matches A)
    const int row  = m0 + col;
    const float f1r = f1[row];
    const float mxr = mx[row];
    const float rsr = rs[row];
    const float* br = bias + (long)row * NN;
    v8f cacc[4] = {v8f{}, v8f{}, v8f{}, v8f{}};
    const int K0 = wave * (NN / 8);

    for (int it = 0; it < (NN / 8) / 32; ++it) {
        const int nb = K0 + it * 32;
        const int n1 = nb + akoff;       // A: K {0..7} or {8..15}
        const int n2 = n1 + 16;          // A: K {16..23} or {24..31}
        v4f bb0 = __builtin_nontemporal_load((const v4f*)(br + n1));
        v4f bb1 = __builtin_nontemporal_load((const v4f*)(br + n1 + 4));
        v4f bb2 = __builtin_nontemporal_load((const v4f*)(br + n2));
        v4f bb3 = __builtin_nontemporal_load((const v4f*)(br + n2 + 4));
        v4f g0 = *(const v4f*)(f2 + n1);
        v4f g1 = *(const v4f*)(f2 + n1 + 4);
        v4f g2 = *(const v4f*)(f2 + n2);
        v4f g3 = *(const v4f*)(f2 + n2 + 4);
        auto coef = [&](float g, float b) -> float {
            float x = fmaxf(f1r + g, 0.f) + b;          // relu(logit) + bias
            return __expf(x - mxr) * rsr + b;           // softmax + bias
        };
        v8u au;
        au[0] = pack2_bf16(coef(g0[0], bb0[0]), coef(g0[1], bb0[1]));
        au[1] = pack2_bf16(coef(g0[2], bb0[2]), coef(g0[3], bb0[3]));
        au[2] = pack2_bf16(coef(g1[0], bb1[0]), coef(g1[1], bb1[1]));
        au[3] = pack2_bf16(coef(g1[2], bb1[2]), coef(g1[3], bb1[3]));
        au[4] = pack2_bf16(coef(g2[0], bb2[0]), coef(g2[1], bb2[1]));
        au[5] = pack2_bf16(coef(g2[2], bb2[2]), coef(g2[3], bb2[3]));
        au[6] = pack2_bf16(coef(g3[0], bb3[0]), coef(g3[1], bb3[1]));
        au[7] = pack2_bf16(coef(g3[2], bb3[2]), coef(g3[3], bb3[3]));
        v16bf a = __builtin_bit_cast(v16bf, au);
#pragma unroll
        for (int t = 0; t < 4; ++t) {
            const int h = t * 16 + col;
            const unsigned short* fp = ftsT + (long)h * NN + nb + bkoff;
            v4u blo = *(const v4u*)(fp);        // K bkoff..bkoff+7
            v4u bhi = *(const v4u*)(fp + 16);   // K bkoff+16..bkoff+23
            v8u bu;
            bu[0] = blo[0]; bu[1] = blo[1]; bu[2] = blo[2]; bu[3] = blo[3];
            bu[4] = bhi[0]; bu[5] = bhi[1]; bu[6] = bhi[2]; bu[7] = bhi[3];
            v16bf b = __builtin_bit_cast(v16bf, bu);
            cacc[t] = __builtin_amdgcn_wmma_f32_16x16x32_bf16(
                false, a, false, b, (short)0, cacc[t], false, false);
        }
    }
    // cross-wave reduction of partial C tiles in LDS
#pragma unroll
    for (int t = 0; t < 4; ++t)
#pragma unroll
        for (int j = 0; j < 8; ++j) {
            int m = j + ((lane < 16) ? 0 : 8);     // C layout: vgpr j = row j / j+8
            int h = t * 16 + col;
            cred[wave * (16 * HH) + m * HH + h] = cacc[t][j];
        }
    __syncthreads();
#pragma unroll
    for (int o = 0; o < 4; ++o) {
        int idx = tid + o * 256;                   // (m, h) in 16 x 64
        float s = 0.f;
#pragma unroll
        for (int w = 0; w < 8; ++w) s += cred[w * (16 * HH) + idx];
        float r = s > 0.f ? s : (__expf(s) - 1.0f);   // elu
        out[(long)(m0 + (idx >> 6)) * HH + (idx & 63)] = r;
    }
}

// ---------------------------------------------------------------------------
extern "C" void kernel_launch(void* const* d_in, const int* in_sizes, int n_in,
                              void* d_out, int out_size, void* d_ws, size_t ws_size,
                              hipStream_t stream) {
    const float* features = (const float*)d_in[0];
    const float* bias     = (const float*)d_in[1];
    const float* W        = (const float*)d_in[2];
    const float* a1       = (const float*)d_in[3];
    const float* b1       = (const float*)d_in[4];
    const float* a2       = (const float*)d_in[5];
    const float* b2       = (const float*)d_in[6];
    float* out = (float*)d_out;

    char* ws = (char*)d_ws;
    unsigned short* ftsT = (unsigned short*)ws;            // 64*8192*2 = 1 MB
    float* f1 = (float*)(ws + (1 << 20));                  // 32 KB each
    float* f2 = f1 + NN;
    float* mx = f2 + NN;
    float* rs = mx + NN;

    proj_kernel<<<NN / 4, 256, 0, stream>>>(features, W, a1, b1, a2, b2, ftsT, f1, f2);
    softstat_kernel<<<NN / 4, 256, 0, stream>>>(bias, f1, f2, mx, rs);
    attn_agg_kernel<<<NN / 16, 256, 0, stream>>>(bias, f1, f2, mx, rs, ftsT, out);
}
